// PredictionLayer_23252952940858
// MI455X (gfx1250) — matrix operand order repaired and checked by reading, not compiled, not run on hardware
//
#include <hip/hip_runtime.h>
#include <stdint.h>

typedef __attribute__((ext_vector_type(16))) __bf16 v16bf;
typedef __attribute__((ext_vector_type(8)))  float  v8f;

#define D_FEAT   512
#define K_DIM    1024            // 2*D_FEAT (concat)
#define HID      1024
#define M_TILE   128             // edges per workgroup
#define N_CHUNK  16              // hidden units per inner pass
#define N_ITERS  (HID / N_CHUNK) // 64
#define A_STRIDE 1032            // K_DIM + 8 bf16 pad (16B) -> bank spread
#define B_STRIDE 1032
#define A_BYTES  ((size_t)M_TILE * A_STRIDE * 2)   // 264192
#define B_BYTES  ((size_t)N_CHUNK * B_STRIDE * 2)  // 33024
#define LDS_BYTES (A_BYTES + B_BYTES + 2 * HID * sizeof(float))  // +b1,W2 stage

static __device__ __forceinline__ unsigned bf16_rne(float f) {
    unsigned u = __float_as_uint(f);
    unsigned r = u + 0x7FFFu + ((u >> 16) & 1u);   // round-to-nearest-even
    return r >> 16;
}
static __device__ __forceinline__ unsigned pack2(float a, float b) {
    return bf16_rne(a) | (bf16_rne(b) << 16);
}

// ---- Kernel 1: W1 [K=1024][N=1024] f32  ->  W1T [N][K] bf16 (workspace) ----
__global__ void w1_to_bf16_T(const float* __restrict__ W1,
                             unsigned short* __restrict__ W1T) {
    __shared__ float tile[32][33];
    const int k0 = blockIdx.x * 32;
    const int n0 = blockIdx.y * 32;
    const int tx = threadIdx.x & 31;
    const int ty = threadIdx.x >> 5;           // 0..7
    #pragma unroll
    for (int i = ty; i < 32; i += 8)
        tile[i][tx] = W1[(size_t)(k0 + i) * HID + (n0 + tx)];
    __syncthreads();
    #pragma unroll
    for (int i = ty; i < 32; i += 8)
        W1T[(size_t)(n0 + i) * K_DIM + (k0 + tx)] =
            (unsigned short)bf16_rne(tile[tx][i]);
}

// ---- Kernel 2: fused gather + GEMM(WMMA bf16) + ReLU + dot(W2) ----
__global__ void __launch_bounds__(256, 1)
edge_mlp_kernel(const float* __restrict__ x,
                const long long* __restrict__ src,
                const long long* __restrict__ dst,
                const unsigned short* __restrict__ W1T,
                const float* __restrict__ b1,
                const float* __restrict__ W2,
                const float* __restrict__ b2,
                float* __restrict__ out,
                int E)
{
    extern __shared__ char smem[];
    unsigned short* sA    = (unsigned short*)smem;                 // [128][A_STRIDE]
    unsigned short* sB    = (unsigned short*)(smem + A_BYTES);     // [16][B_STRIDE]
    float*          sBias = (float*)(smem + A_BYTES + B_BYTES);    // [1024]
    float*          sW2   = sBias + HID;                           // [1024]

    const int tid   = threadIdx.x;
    const int tile0 = blockIdx.x * M_TILE;

    // ---- stage b1 / W2 into LDS (once) ----
    {
        const float4* g1 = (const float4*)b1;
        const float4* g2 = (const float4*)W2;
        float4 v1 = g1[tid];
        float4 v2 = g2[tid];
        *(float4*)&sBias[tid * 4] = v1;
        *(float4*)&sW2[tid * 4]   = v2;
    }

    // ---- gather phase: 2 threads per edge (src half / dst half), f32->bf16 ----
    {
        const int m    = tid >> 1;
        const int half = tid & 1;
        const int e    = tile0 + m;
        unsigned* dp = (unsigned*)&sA[m * A_STRIDE + half * D_FEAT];
        if (e < E) {
            const long long idx = half ? dst[e] : src[e];
            const float4* rowp = (const float4*)(x + (size_t)idx * D_FEAT);
            #pragma unroll 4
            for (int j = 0; j < D_FEAT / 4; ++j) {
                float4 v = rowp[j];
                dp[2 * j]     = pack2(v.x, v.y);
                dp[2 * j + 1] = pack2(v.z, v.w);
            }
        } else {
            #pragma unroll 4
            for (int j = 0; j < D_FEAT / 2; ++j) dp[j] = 0u;
        }
    }

    const int wave = tid >> 5;
    const int lane = tid & 31;
    const int hi   = lane >> 4;          // half-wave select
    const int l15  = lane & 15;
    const int m0   = wave * 16;          // each wave owns 16 edges

    float sacc[8];
    #pragma unroll
    for (int r = 0; r < 8; ++r) sacc[r] = 0.f;

    // A operand (16-bit 16x32 layout): lanes 0-15 take K k+{0..7,16..23},
    // lanes 16-31 take K k+{8..15,24..31}, row M = m0 + (lane&15).
    const unsigned short* aBase = &sA[(m0 + l15) * A_STRIDE + hi * 8];
    // B operand: lane column N = lane&15; lanes 0-15 K k+{0..7,16..23}, lanes 16-31 K k+{8..15,24..31}.
    const unsigned short* bBase = &sB[l15 * B_STRIDE + hi * 8];

    // per-thread loop-invariant B staging slots (16 rows x 128 uint4)
    unsigned short* bStore[8];
    #pragma unroll
    for (int t = 0; t < 8; ++t) {
        const int idx = tid + t * 256;     // 0..2047
        bStore[t] = &sB[(idx >> 7) * B_STRIDE + (idx & 127) * 8];
    }

    // prefetch chunk 0 of B into registers (overlaps with gather stores above)
    uint4 breg[8];
    {
        const uint4* g = (const uint4*)W1T;
        #pragma unroll
        for (int t = 0; t < 8; ++t) breg[t] = g[tid + t * 256];
    }

    __syncthreads();   // gather + bias staging complete

    for (int nc = 0; nc < N_ITERS; ++nc) {
        // commit staged B registers to LDS
        #pragma unroll
        for (int t = 0; t < 8; ++t) *(uint4*)bStore[t] = breg[t];
        __syncthreads();

        // software pipeline: issue next chunk's global loads now, consume next iter
        if (nc + 1 < N_ITERS) {
            const uint4* g = (const uint4*)(W1T + (size_t)(nc + 1) * N_CHUNK * K_DIM);
            #pragma unroll
            for (int t = 0; t < 8; ++t) breg[t] = g[tid + t * 256];
        }

        v8f c = {};
        #pragma unroll 4
        for (int kk = 0; kk < K_DIM / 32; ++kk) {
            const int k = kk * 32;
            union { uint4 u[2]; v16bf v; } a, b;
            a.u[0] = *(const uint4*)(aBase + k);
            a.u[1] = *(const uint4*)(aBase + k + 16);
            b.u[0] = *(const uint4*)(bBase + k);
            b.u[1] = *(const uint4*)(bBase + k + 16);
            c = __builtin_amdgcn_wmma_f32_16x16x32_bf16(
                    /*neg_a=*/false, a.v, /*neg_b=*/false, b.v,
                    /*c_mod=*/(short)0, c, /*reuse_a=*/false, /*reuse_b=*/false);
        }

        // fused epilogue: h = relu(c + b1[n]); score += h * W2[n]  (LDS-resident)
        const int   n   = nc * N_CHUNK + l15;
        const float b1v = sBias[n];
        const float w2v = sW2[n];
        #pragma unroll
        for (int r = 0; r < 8; ++r) {
            float h = c[r] + b1v;
            h = fmaxf(h, 0.f);
            sacc[r] = fmaf(h, w2v, sacc[r]);
        }
        __syncthreads();   // all waves done reading sB before next commit
    }

    // reduce over N (the 16 lanes of each half-wave)
    #pragma unroll
    for (int off = 8; off >= 1; off >>= 1) {
        #pragma unroll
        for (int r = 0; r < 8; ++r)
            sacc[r] += __shfl_xor(sacc[r], off, 32);
    }

    // D layout: c[r] holds element (M = r + 8*hi, N = lane&15)
    if (l15 == 0) {
        const float bias2 = b2[0];
        const int mb = m0 + hi * 8;
        #pragma unroll
        for (int r = 0; r < 8; ++r) {
            const int e = tile0 + mb + r;
            if (e < E) out[e] = sacc[r] + bias2;
        }
    }
}

extern "C" void kernel_launch(void* const* d_in, const int* in_sizes, int n_in,
                              void* d_out, int out_size, void* d_ws, size_t ws_size,
                              hipStream_t stream) {
    const float*      x   = (const float*)d_in[0];
    const long long*  src = (const long long*)d_in[1];
    const long long*  dst = (const long long*)d_in[2];
    const float*      W1  = (const float*)d_in[3];
    const float*      b1  = (const float*)d_in[4];
    const float*      W2  = (const float*)d_in[5];
    const float*      b2  = (const float*)d_in[6];
    float* out = (float*)d_out;
    const int E = in_sizes[1];

    unsigned short* W1T = (unsigned short*)d_ws;   // 1024*1024*2 = 2 MB scratch

    dim3 tgrid(K_DIM / 32, HID / 32);
    w1_to_bf16_T<<<tgrid, 256, 0, stream>>>(W1, W1T);

    const int ntiles = (E + M_TILE - 1) / M_TILE;
    edge_mlp_kernel<<<ntiles, 256, LDS_BYTES, stream>>>(x, src, dst, W1T, b1,
                                                        W2, b2, out, E);
}
